// NodeClassifierGNN_conv_19078244729017
// MI455X (gfx1250) — compile-verified, matrix operand order checked by reading, not run on hardware
//
#include <hip/hip_runtime.h>
#include <hip/hip_bf16.h>
#include <stdint.h>

typedef __attribute__((ext_vector_type(16))) _Float16 v16h;
typedef __attribute__((ext_vector_type(8)))  float    v8f;

#define GNN_N   50000
#define GNN_E   200000
#define GNN_EIN 154
#define GNN_KP  160   // 154 padded to 5 K-panels of 32
#define GNN_H   32

// ---------------------------------------------------------------------------
// Wave-internal LDS fence: DS ops are in-order within a wave; this stops the
// compiler hoisting cross-lane LDS loads above stores and retires DS ops.
// ---------------------------------------------------------------------------
__device__ __forceinline__ void wave_lds_sync() {
#if __has_builtin(__builtin_amdgcn_s_wait_dscnt)
    __builtin_amdgcn_s_wait_dscnt(0);
#else
    asm volatile("s_wait_dscnt 0" ::: "memory");
#endif
    __builtin_amdgcn_wave_barrier();
}

__device__ __forceinline__ uint32_t pack_f16x2(float a, float b) {
#if __has_builtin(__builtin_amdgcn_cvt_pkrtz)
    auto p = __builtin_amdgcn_cvt_pkrtz(a, b);   // __fp16 ext_vector(2)
    return __builtin_bit_cast(uint32_t, p);
#else
    union { _Float16 h[2]; uint32_t u; } c;
    c.h[0] = (_Float16)a; c.h[1] = (_Float16)b; return c.u;
#endif
}

union Frag16 { v16h h; uint4 q[2]; };

// One f16 WMMA operand (A 16x32 row, or B 32x16 column) = two 16B reads.
// baseU32 points at this lane's row/column start. Per ISA 7.12.2 the lane's
// (K,K+1) dword pairs for VGPRs v0..3 sit at u32 offsets {0..3}+4g and for
// v4..7 at {8..11}+4g, g = lane>>4  ->  uint4 indices g and 2+g.
__device__ __forceinline__ v16h load_frag_pair(const uint32_t* baseU32, int g) {
    const uint4* p = (const uint4*)baseU32;
    Frag16 f;
    f.q[0] = p[g];
    f.q[1] = p[2 + g];
    return f.h;
}

// ---------------------------------------------------------------------------
// Fused NNConv edge kernel: one wave per 16-edge tile.
// FIRST=true  : h = relu(EA@W1a+b1a) via WMMA over 5 K-panels; additionally
//               computes h2 = relu(EA@W1b+b1b) (next layer's MLP1) from the
//               same A fragments and spills it as f16 [E,32] -> hIn of layer 2
//               never re-reads the 123 MB edge_attr array.
// FIRST=false : A operand loaded straight from hIn with 2 global b128 reads.
// Then: msg = sum_i xs[:,i] * (h @ W2[:,32i:32i+32] + b2_chunk), chunk WMMAs,
// scatter-add with f32 global atomics.
// ---------------------------------------------------------------------------
template <int CIN, bool FIRST>
__global__ __launch_bounds__(256) void edge_nnconv_kernel(
    const float*    __restrict__ x,      // [N, CIN]
    const int*      __restrict__ src,    // [E]
    const int*      __restrict__ dst,    // [E]
    const float*    __restrict__ ea,     // [E, 154]        (FIRST only)
    const float*    __restrict__ W1a,    // [154, 32]       (FIRST only)
    const float*    __restrict__ b1a,    // [32]            (FIRST only)
    const float*    __restrict__ W1b,    // [154, 32]       (FIRST only)
    const float*    __restrict__ b1b,    // [32]            (FIRST only)
    const float*    __restrict__ W2,     // [32, CIN*32]
    const float*    __restrict__ b2,     // [CIN*32]
    const _Float16* __restrict__ hIn,    // [E, 32] f16     (!FIRST only)
    _Float16*       __restrict__ h2out,  // [E, 32] f16     (FIRST only)
    float*          __restrict__ agg,    // [N, 32]
    int E)
{
    constexpr int NC = CIN * 32;
    extern __shared__ char smem[];
    _Float16* ldsW1a = (_Float16*)smem;                 // [32 cols][160 k] f16
    _Float16* ldsW1b = (_Float16*)(smem + 10240);       // [32 cols][160 k] f16
    float*    ldsB1a = (float*)(smem + 20480);          // [32]
    float*    ldsB1b = (float*)(smem + 20608);          // [32]
    _Float16* ldsW2  = (_Float16*)(smem + 20736);       // [NC cols][32 k] f16
    float*    ldsB2  = (float*)(smem + 20736 + NC * 64);// [NC]
    char*     waveMem = smem + 20736 + NC * 64 + NC * 4;
    constexpr int PER_WAVE = 1024 + 16 * CIN * 4 + 64;

    const int tid    = threadIdx.x;
    const int lane   = tid & 31;
    const int waveId = tid >> 5;

    _Float16* aTile = (_Float16*)(waveMem + waveId * PER_WAVE);  // [16][32] f16
    float*    xsT   = (float*)((char*)aTile + 1024);             // [16][CIN]
    int*      dIdx  = (int*)((char*)xsT + 16 * CIN * 4);         // [16]

    // ---- cooperative weight staging (block-wide, f32 -> f16, column-major) ----
    if constexpr (FIRST) {
        for (int idx = tid; idx < 32 * GNN_KP; idx += 256) {
            int c = idx / GNN_KP, k = idx - c * GNN_KP;
            ldsW1a[idx] = (_Float16)((k < GNN_EIN) ? W1a[k * 32 + c] : 0.f);
            ldsW1b[idx] = (_Float16)((k < GNN_EIN) ? W1b[k * 32 + c] : 0.f);
        }
        if (tid < 32) { ldsB1a[tid] = b1a[tid]; ldsB1b[tid] = b1b[tid]; }
    }
    for (int idx = tid; idx < NC * 32; idx += 256) {
        int c = idx >> 5, k = idx & 31;
        ldsW2[idx] = (_Float16)W2[k * NC + c];
    }
    for (int idx = tid; idx < NC; idx += 256) ldsB2[idx] = b2[idx];
    __syncthreads();

    const int numTiles = E >> 4;            // E % 16 == 0 here
    int tile = blockIdx.x * 8 + waveId;
    const bool active = tile < numTiles;
    if (!active) tile = numTiles - 1;       // idle waves shadow last tile, outputs masked
    const int e0 = tile << 4;

    const int n = lane & 15;                // output column within 16-wide half
    const int g = lane >> 4;                // lane half (selects fragment K-range)

    // ---- gather dst indices + source node features into LDS ----
    {
        const int r = n;
        const int e = e0 + r;
        const int s = src[e];
        if (g == 0) dIdx[r] = dst[e];
        if constexpr ((CIN & 3) == 0) {     // aligned rows: vector gather
            const float4* xr = (const float4*)(x + (size_t)s * CIN);
            float4* xw = (float4*)(xsT + r * CIN);
            for (int q = g; q < CIN / 4; q += 2) xw[q] = xr[q];
        } else {
            for (int i = g; i < CIN; i += 2)
                xsT[r * CIN + i] = x[(size_t)s * CIN + i];
        }
    }
    wave_lds_sync();

    uint32_t* aU32 = (uint32_t*)aTile;
    v16h ha;                                 // A operand of the chunk GEMM (h, f16)

    if constexpr (FIRST) {
        // ---- h  = relu(EA @ W1a + b1a)  and  h2 = relu(EA @ W1b + b1b) ----
        v8f h0, h1, p0, p1;
        {
            const float ca = ldsB1a[n], cb = ldsB1a[16 + n];
            const float da = ldsB1b[n], db = ldsB1b[16 + n];
#pragma unroll
            for (int j = 0; j < 8; ++j) { h0[j] = ca; h1[j] = cb; p0[j] = da; p1[j] = db; }
        }
        const uint32_t* w1au = (const uint32_t*)ldsW1a;   // col stride 80 u32
        const uint32_t* w1bu = (const uint32_t*)ldsW1b;
#pragma unroll
        for (int ks = 0; ks < 5; ++ks) {
            // stage 16x32 EA panel as f16: 8 float2 loads (rows are 8B-aligned,
            // 154 is even so each float2 is fully valid or fully padded)
            float2 vb[8];
#pragma unroll
            for (int t = 0; t < 8; ++t) {
                const int s4 = lane + 32 * t;
                const int r = s4 >> 4, q = s4 & 15;
                const int K0 = ks * 32 + q * 2;
                float2 v; v.x = 0.f; v.y = 0.f;
                if (K0 < GNN_EIN)
                    v = *(const float2*)(ea + (size_t)(e0 + r) * GNN_EIN + K0);
                vb[t] = v;
            }
#pragma unroll
            for (int t = 0; t < 8; ++t) {
                const int s4 = lane + 32 * t;
                const int r = s4 >> 4, q = s4 & 15;
                aU32[r * 16 + q] = pack_f16x2(vb[t].x, vb[t].y);
            }
            wave_lds_sync();
            v16h a   = load_frag_pair(aU32 + n * 16, g);  // row = lane&15
            v16h bL  = load_frag_pair(w1au + n * 80 + ks * 16, g);
            v16h bH  = load_frag_pair(w1au + (16 + n) * 80 + ks * 16, g);
            h0 = __builtin_amdgcn_wmma_f32_16x16x32_f16(false, a, false, bL, (short)0, h0, false, false);
            h1 = __builtin_amdgcn_wmma_f32_16x16x32_f16(false, a, false, bH, (short)0, h1, false, false);
            v16h bL2 = load_frag_pair(w1bu + n * 80 + ks * 16, g);
            v16h bH2 = load_frag_pair(w1bu + (16 + n) * 80 + ks * 16, g);
            p0 = __builtin_amdgcn_wmma_f32_16x16x32_f16(false, a, false, bL2, (short)0, p0, false, false);
            p1 = __builtin_amdgcn_wmma_f32_16x16x32_f16(false, a, false, bH2, (short)0, p1, false, false);
            wave_lds_sync();
        }
#pragma unroll
        for (int j = 0; j < 8; ++j) {
            h0[j] = fmaxf(h0[j], 0.f); h1[j] = fmaxf(h1[j], 0.f);
            p0[j] = fmaxf(p0[j], 0.f); p1[j] = fmaxf(p1[j], 0.f);
        }

        // restage h (C layout -> row-major f16), pick up as A operand
#pragma unroll
        for (int j = 0; j < 8; ++j) {
            const int r = j + 8 * g;       // C layout: VGPR j holds row j (+8 upper lanes)
            aTile[r * 32 + n]      = (_Float16)h0[j];
            aTile[r * 32 + 16 + n] = (_Float16)h1[j];
        }
        wave_lds_sync();
        ha = load_frag_pair(aU32 + n * 16, g);
        wave_lds_sync();

        // restage h2 and spill tile (contiguous 1 KB) for layer 2
#pragma unroll
        for (int j = 0; j < 8; ++j) {
            const int r = j + 8 * g;
            aTile[r * 32 + n]      = (_Float16)p0[j];
            aTile[r * 32 + 16 + n] = (_Float16)p1[j];
        }
        wave_lds_sync();
        if (active) {
            const uint4* a4 = (const uint4*)aTile;
            uint4* o4 = (uint4*)(h2out + (size_t)e0 * 32);
            o4[lane]      = a4[lane];
            o4[lane + 32] = a4[lane + 32];
        }
    } else {
        // A operand straight from the f16 spill: two global b128 reads
        const uint32_t* hrow = (const uint32_t*)(hIn + (size_t)e0 * 32);
        ha = load_frag_pair(hrow + n * 16, g);
    }

    // ---- msg = sum_i xs[:,i] * (h @ W2[:,32i:32i+32] + b2_chunk) ----
    v8f m0, m1;
#pragma unroll
    for (int j = 0; j < 8; ++j) { m0[j] = 0.f; m1[j] = 0.f; }
    const uint32_t* w2u = (const uint32_t*)ldsW2;   // col stride 16 u32
#pragma unroll 4
    for (int i = 0; i < CIN; ++i) {
        v16h bL = load_frag_pair(w2u + (i * 32 + n) * 16, g);
        v16h bH = load_frag_pair(w2u + (i * 32 + 16 + n) * 16, g);
        const float c0s = ldsB2[i * 32 + n];
        const float c1s = ldsB2[i * 32 + 16 + n];
        v8f c0, c1;
#pragma unroll
        for (int j = 0; j < 8; ++j) { c0[j] = c0s; c1[j] = c1s; }
        c0 = __builtin_amdgcn_wmma_f32_16x16x32_f16(false, ha, false, bL, (short)0, c0, false, false);
        c1 = __builtin_amdgcn_wmma_f32_16x16x32_f16(false, ha, false, bH, (short)0, c1, false, false);
#pragma unroll
        for (int j = 0; j < 8; ++j) {
            const float xv = xsT[(j + 8 * g) * CIN + i];
            m0[j] = fmaf(xv, c0[j], m0[j]);
            m1[j] = fmaf(xv, c1[j], m1[j]);
        }
    }

    // ---- scatter-mean numerator: agg[dst] += msg ----
    if (active) {
#pragma unroll
        for (int j = 0; j < 8; ++j) {
            const int r = j + 8 * g;
            const int d = dIdx[r];
            atomicAdd(&agg[(size_t)d * 32 + n],      m0[j]);
            atomicAdd(&agg[(size_t)d * 32 + 16 + n], m1[j]);
        }
    }
}

// ---------------------------------------------------------------------------
// Small supporting kernels (bandwidth-trivial)
// ---------------------------------------------------------------------------
__global__ void zero_kernel(float* __restrict__ p, long n) {
    long i = (long)blockIdx.x * blockDim.x + threadIdx.x;
    if (i < n) p[i] = 0.f;
}

__global__ void deg_kernel(const int* __restrict__ dst, float* __restrict__ deg, int E) {
    int e = blockIdx.x * blockDim.x + threadIdx.x;
    if (e < E) atomicAdd(&deg[dst[e]], 1.0f);
}

template <int CIN>
__global__ void node_update_kernel(const float* __restrict__ agg, const float* __restrict__ deg,
                                   const float* __restrict__ xin, const float* __restrict__ root,
                                   const float* __restrict__ bias, float* __restrict__ xout, int N) {
    int t = blockIdx.x * blockDim.x + threadIdx.x;
    if (t >= N * 32) return;
    int node = t >> 5, o = t & 31;
    float s = agg[(size_t)node * 32 + o] / fmaxf(deg[node], 1.0f) + bias[o];
#pragma unroll 4
    for (int i = 0; i < CIN; ++i) s = fmaf(xin[(size_t)node * CIN + i], root[i * 32 + o], s);
    xout[t] = fmaxf(s, 0.f);
}

__global__ void head_kernel(const float* __restrict__ x2, const float* __restrict__ lw,
                            const float* __restrict__ lb, float* __restrict__ out, int N) {
    int nd = blockIdx.x * blockDim.x + threadIdx.x;
    if (nd >= N) return;
    float s = lb[0];
#pragma unroll
    for (int i = 0; i < 32; ++i) s = fmaf(x2[(size_t)nd * 32 + i], lw[i], s);
    out[nd] = s;
}

static inline size_t edge_lds_bytes(int CIN) {
    size_t NC = (size_t)CIN * 32;
    return 20736 + NC * 64 + NC * 4
         + 8 * (1024 + (size_t)16 * CIN * 4 + 64);
}

extern "C" void kernel_launch(void* const* d_in, const int* in_sizes, int n_in,
                              void* d_out, int out_size, void* d_ws, size_t ws_size,
                              hipStream_t stream) {
    constexpr int N = GNN_N, E = GNN_E;
    const float* x     = (const float*)d_in[0];
    const int*   eidx  = (const int*)  d_in[1];
    const float* ea    = (const float*)d_in[2];
    const float* e1W1  = (const float*)d_in[3];
    const float* e1b1  = (const float*)d_in[4];
    const float* e1W2  = (const float*)d_in[5];
    const float* e1b2  = (const float*)d_in[6];
    const float* root1 = (const float*)d_in[7];
    const float* bias1 = (const float*)d_in[8];
    const float* e2W1  = (const float*)d_in[9];
    const float* e2b1  = (const float*)d_in[10];
    const float* e2W2  = (const float*)d_in[11];
    const float* e2b2  = (const float*)d_in[12];
    const float* root2 = (const float*)d_in[13];
    const float* bias2 = (const float*)d_in[14];
    const float* linW  = (const float*)d_in[15];
    const float* linb  = (const float*)d_in[16];

    const int* src  = eidx;
    const int* dstp = eidx + E;

    // workspace: x1 | agg | deg | x2 | h2(f16)   (~32 MB total)
    float*     x1  = (float*)d_ws;
    float*     agg = x1 + (size_t)N * 32;
    float*     deg = agg + (size_t)N * 32;
    float*     x2  = deg + N;
    _Float16*  h2  = (_Float16*)(x2 + (size_t)N * 32);   // [E,32] f16

    {   // zero agg + deg (adjacent in workspace)
        long n = (long)N * 32 + N;
        zero_kernel<<<(int)((n + 255) / 256), 256, 0, stream>>>(agg, n);
    }
    deg_kernel<<<(E + 255) / 256, 256, 0, stream>>>(dstp, deg, E);

    const int numTiles = E / 16;
    const int blocks   = (numTiles + 7) / 8;   // 8 waves (tiles) per 256-thread block

    // Layer 1 (also produces h2 = relu(ea @ e2_W1 + e2_b1) so edge_attr is read once)
    edge_nnconv_kernel<15, true><<<blocks, 256, edge_lds_bytes(15), stream>>>(
        x, src, dstp, ea, e1W1, e1b1, e2W1, e2b1, e1W2, e1b2,
        (const _Float16*)nullptr, h2, agg, E);
    node_update_kernel<15><<<(N * 32 + 255) / 256, 256, 0, stream>>>(
        agg, deg, x, root1, bias1, x1, N);

    {   // re-zero agg for layer 2
        long n = (long)N * 32;
        zero_kernel<<<(int)((n + 255) / 256), 256, 0, stream>>>(agg, n);
    }
    edge_nnconv_kernel<32, false><<<blocks, 256, edge_lds_bytes(32), stream>>>(
        x1, src, dstp, ea, nullptr, nullptr, nullptr, nullptr, e2W2, e2b2,
        h2, (_Float16*)nullptr, agg, E);
    node_update_kernel<32><<<(N * 32 + 255) / 256, 256, 0, stream>>>(
        agg, deg, x1, root2, bias2, x2, N);

    head_kernel<<<(N + 255) / 256, 256, 0, stream>>>(x2, linW, linb, (float*)d_out, N);
}
